// SourceExpertMoE_17188459119127
// MI455X (gfx1250) — matrix operand order repaired and checked by reading, not compiled, not run on hardware
//
#include <hip/hip_runtime.h>
#include <hip/hip_bf16.h>

typedef unsigned short u16;
typedef unsigned int   u32;
typedef __attribute__((ext_vector_type(16))) __bf16 v16bf;
typedef __attribute__((ext_vector_type(8)))  float  v8f;

#define MM 8
#define NN 4096
#define DD 512
#define HH 2048
#define OO 512
#define TOPK 2
#define NSLOT 8448      // 8192 dispatched + per-expert padding to 32
#define MAXTILE 264     // NSLOT / 32
#define PITCH 48        // LDS pitch in u16 (96 B, 16B-aligned rows)

// ---------- helpers ----------
// Native f32 -> bf16 convert (RNE); lets the compiler use v_cvt_*bf16* ops
// instead of a 4-VALU bit-twiddle per element.
__device__ __forceinline__ u16 f2bf(float f) {
  union { __bf16 b; u16 u; } t;
  t.b = (__bf16)f;
  return t.u;
}
__device__ __forceinline__ float bf2f(u16 h) {
  return __uint_as_float(((u32)h) << 16);
}

union V16U { uint4 q[2]; u16 h[16]; v16bf v; };
union V4U  { uint2 u; u16 h[4]; };

// A fragment: 16x32 bf16, row-major tile in LDS [row][k].
// lane<16: row=row0+lane, K elems {0..7,16..23}; lane>=16: K {8..15,24..31}.
__device__ __forceinline__ v16bf frag_a(const u16* lds, int row0, int lane) {
  int row  = row0 + (lane & 15);
  int base = (lane & 16) ? 8 : 0;
  const u16* p = lds + row * PITCH + base;
  V16U t;
  t.q[0] = *(const uint4*)(p);
  t.q[1] = *(const uint4*)(p + 16);
  return t.v;
}
// B fragment: 32x16 bf16, transposed tile in LDS [n][k].
// lane<16: n=col0+lane, K elems {0..7,16..23}; lane>=16: K {8..15,24..31}.
__device__ __forceinline__ v16bf frag_b(const u16* lds, int col0, int lane) {
  int col  = col0 + (lane & 15);
  int koff = (lane & 16) ? 8 : 0;
  const u16* p = lds + col * PITCH + koff;
  V16U t;
  t.q[0] = *(const uint4*)(p);
  t.q[1] = *(const uint4*)(p + 16);
  return t.v;
}

// Load 4 consecutive k-rows x 4 cols of a row-major f32 matrix.
__device__ __forceinline__ void loadB4x4(const float* W, int ldw, int k, int c, float4* b) {
  const float* src = W + (size_t)k * ldw + c;
  #pragma unroll
  for (int i = 0; i < 4; i++) b[i] = *(const float4*)(src + (size_t)i * ldw);
}
// Transpose 4x4 in registers, pack to bf16, store 4x uint2 into Bs[col][k].
__device__ __forceinline__ void storeBT4x4(u16* Bs, int kk, int cc, const float4* b) {
  #pragma unroll
  for (int j = 0; j < 4; j++) {
    V4U t;
    t.h[0] = f2bf((&b[0].x)[j]);
    t.h[1] = f2bf((&b[1].x)[j]);
    t.h[2] = f2bf((&b[2].x)[j]);
    t.h[3] = f2bf((&b[3].x)[j]);
    *(uint2*)(Bs + (cc + j) * PITCH + kk) = t.u;
  }
}
// Pack 16 f32 -> 16 bf16 and store as 2x uint4.
__device__ __forceinline__ void storeA16(u16* As, int row, int koff, const float4* a) {
  V16U t;
  #pragma unroll
  for (int i = 0; i < 4; i++) {
    t.h[4*i+0] = f2bf((&a[i].x)[0]);
    t.h[4*i+1] = f2bf((&a[i].x)[1]);
    t.h[4*i+2] = f2bf((&a[i].x)[2]);
    t.h[4*i+3] = f2bf((&a[i].x)[3]);
  }
  uint4* dst = (uint4*)(As + row * PITCH + koff);
  dst[0] = t.q[0]; dst[1] = t.q[1];
}

// ---------- kernel 0: zero init ----------
__global__ __launch_bounds__(256) void k_zero(float* __restrict__ outF,
                                              float* __restrict__ outTot,
                                              int* __restrict__ counts) {
  size_t i = (size_t)blockIdx.x * 256 + threadIdx.x;
  if (i < (size_t)NN * OO) outF[i] = 0.f;
  if (i < MM) { outTot[i] = 0.f; counts[i] = 0; }
}

// ---------- kernel 1: gateH = relu(fs @ Wg1 + bg1), bf16 out ----------
// fs[n][k] = features[(k>>9)*NN + n][k&511]
__global__ __launch_bounds__(256) void k_gate1(const float* __restrict__ features,
                                               const float* __restrict__ Wg1,
                                               const float* __restrict__ bg1,
                                               u16* __restrict__ gateH) {
  __shared__ u16 As[2 * 128 * PITCH];
  __shared__ u16 Bs[2 * 128 * PITCH];
  const int tid = threadIdx.x, lane = tid & 31, wave = tid >> 5;
  const int n0 = blockIdx.x * 128, h0 = blockIdx.y * 128;
  const int wr = (wave >> 1) * 32;     // wave row base (0,32,64,96)
  const int wc = (wave & 1) * 64;      // wave col base (0,64)

  v8f acc[2][4];
  #pragma unroll
  for (int i = 0; i < 2; i++)
    #pragma unroll
    for (int j = 0; j < 4; j++) {
      v8f z = {0.f,0.f,0.f,0.f,0.f,0.f,0.f,0.f};
      acc[i][j] = z;
    }

  const int ar = tid >> 1, ah = (tid & 1) * 16;     // A stage: 128 rows x 32 k
  const int bkg = (tid >> 5) * 4, bcg = (tid & 31) * 4; // B stage: 4x4 blocks

  const int KSTEPS = (MM * DD) / 32;
  float4 ra[4], rb[4];
  // prologue: load step 0 into registers
  {
    int k = 0 + ah;
    const float* src = features + ((size_t)(k >> 9) * NN + (n0 + ar)) * DD + (k & 511);
    #pragma unroll
    for (int i = 0; i < 4; i++) ra[i] = ((const float4*)src)[i];
    loadB4x4(Wg1, HH, 0 + bkg, h0 + bcg, rb);
  }
  for (int s = 0; s < KSTEPS; s++) {
    u16* Ab = As + (s & 1) * (128 * PITCH);
    u16* Bb = Bs + (s & 1) * (128 * PITCH);
    storeA16(Ab, ar, ah, ra);
    storeBT4x4(Bb, bkg, bcg, rb);
    __syncthreads();
    if (s + 1 < KSTEPS) {   // overlap next global load with WMMA on this buffer
      int k = (s + 1) * 32 + ah;
      const float* srcA = features + ((size_t)(k >> 9) * NN + (n0 + ar)) * DD + (k & 511);
      #pragma unroll
      for (int i = 0; i < 4; i++) ra[i] = ((const float4*)srcA)[i];
      loadB4x4(Wg1, HH, (s + 1) * 32 + bkg, h0 + bcg, rb);
      if (s + 2 < KSTEPS) {
        int k2 = (s + 2) * 32 + ah;
        __builtin_prefetch(features + ((size_t)(k2 >> 9) * NN + (n0 + ar)) * DD + (k2 & 511), 0, 1);
        __builtin_prefetch(Wg1 + (size_t)((s + 2) * 32 + bkg) * HH + h0 + bcg, 0, 1);
      }
    }
    v16bf afr[2], bfr[4];
    #pragma unroll
    for (int mi = 0; mi < 2; mi++) afr[mi] = frag_a(Ab, wr + mi * 16, lane);
    #pragma unroll
    for (int ni = 0; ni < 4; ni++) bfr[ni] = frag_b(Bb, wc + ni * 16, lane);
    #pragma unroll
    for (int mi = 0; mi < 2; mi++)
      #pragma unroll
      for (int ni = 0; ni < 4; ni++)
        acc[mi][ni] = __builtin_amdgcn_wmma_f32_16x16x32_bf16(
            false, afr[mi], false, bfr[ni], (short)0, acc[mi][ni], false, false);
  }
  // epilogue: bias + relu + bf16 store
  #pragma unroll
  for (int mi = 0; mi < 2; mi++) {
    int rbase = n0 + wr + mi * 16 + ((lane & 16) ? 8 : 0);
    #pragma unroll
    for (int ni = 0; ni < 4; ni++) {
      int h = h0 + wc + ni * 16 + (lane & 15);
      float b = bg1[h];
      #pragma unroll
      for (int j = 0; j < 8; j++) {
        float v = acc[mi][ni][j] + b;
        v = v > 0.f ? v : 0.f;
        gateH[(size_t)(rbase + j) * HH + h] = f2bf(v);
      }
    }
  }
}

// ---------- kernel 2: gate logits, softmax, +receptivity, top-2, ranks ----------
__global__ __launch_bounds__(256) void k_gate2(const u16* __restrict__ gateH,
                                               const float* __restrict__ Wg2,
                                               const float* __restrict__ bg2,
                                               const float* __restrict__ recept,
                                               int* __restrict__ top_i,
                                               float* __restrict__ top_w,
                                               int* __restrict__ counts,
                                               float* __restrict__ outRanks,
                                               float* __restrict__ outTot) {
  const int n = blockIdx.x;
  const int tid = threadIdx.x, lane = tid & 31, wave = tid >> 5;
  float a[8] = {0.f,0.f,0.f,0.f,0.f,0.f,0.f,0.f};
  for (int j = tid; j < HH; j += 256) {
    float g = bf2f(gateH[(size_t)n * HH + j]);
    const float4* w = (const float4*)(Wg2 + (size_t)j * MM);
    float4 w0 = w[0], w1 = w[1];
    a[0] += g * w0.x; a[1] += g * w0.y; a[2] += g * w0.z; a[3] += g * w0.w;
    a[4] += g * w1.x; a[5] += g * w1.y; a[6] += g * w1.z; a[7] += g * w1.w;
  }
  #pragma unroll
  for (int off = 16; off > 0; off >>= 1)
    #pragma unroll
    for (int m = 0; m < 8; m++) a[m] += __shfl_down(a[m], off);
  __shared__ float red[8][8];
  if (lane == 0)
    #pragma unroll
    for (int m = 0; m < 8; m++) red[wave][m] = a[m];
  __syncthreads();
  if (tid == 0) {
    float lg[8];
    #pragma unroll
    for (int m = 0; m < 8; m++) {
      float s = bg2[m];
      #pragma unroll
      for (int w = 0; w < 8; w++) s += red[w][m];
      lg[m] = s;
    }
    float mx = lg[0];
    #pragma unroll
    for (int m = 1; m < 8; m++) mx = lg[m] > mx ? lg[m] : mx;
    float ex[8], sum = 0.f;
    #pragma unroll
    for (int m = 0; m < 8; m++) { ex[m] = __expf(lg[m] - mx); sum += ex[m]; }
    float sc[8];
    #pragma unroll
    for (int m = 0; m < 8; m++) sc[m] = ex[m] / sum + recept[(size_t)n * MM + m];
    int i1 = 0; float v1 = sc[0];
    #pragma unroll
    for (int m = 1; m < 8; m++) if (sc[m] > v1) { v1 = sc[m]; i1 = m; }
    int i2 = -1; float v2 = -1e30f;
    #pragma unroll
    for (int m = 0; m < 8; m++) if (m != i1 && sc[m] > v2) { v2 = sc[m]; i2 = m; }
    float inv = 1.f / (v1 + v2);
    top_i[2 * n] = i1; top_i[2 * n + 1] = i2;
    top_w[2 * n] = v1 * inv; top_w[2 * n + 1] = v2 * inv;
    atomicAdd(&counts[i1], 1); atomicAdd(&counts[i2], 1);
    #pragma unroll
    for (int m = 0; m < 8; m++) {
      int r = (m == i1) ? 0 : ((m == i2) ? 1 : TOPK);
      outRanks[(size_t)m * NN + n] = (float)r;
      unsafeAtomicAdd(&outTot[m], (float)r);
    }
  }
}

// ---------- kernel 3: scan counts -> offsets / tile table; init slots ----------
__global__ __launch_bounds__(256) void k_scan(const int* __restrict__ counts,
                                              int* __restrict__ offsets,
                                              int* __restrict__ cursors,
                                              int* __restrict__ tile_e,
                                              int* __restrict__ tile_s0,
                                              int* __restrict__ slot_token) {
  const int tid = threadIdx.x;
  for (int s = tid; s < NSLOT; s += 256) slot_token[s] = -1;
  if (tid == 0) {
    int off = 0, t = 0;
    for (int e = 0; e < MM; e++) {
      offsets[e] = off;
      cursors[e] = 0;
      int pc = ((counts[e] + 31) >> 5) << 5;
      int nt = pc >> 5;
      for (int k = 0; k < nt; k++)
        if (t < MAXTILE) { tile_e[t] = e; tile_s0[t] = off + k * 32; t++; }
      off += pc;
    }
    offsets[MM] = off;
    for (; t < MAXTILE; t++) { tile_e[t] = -1; tile_s0[t] = 0; }
  }
}

// ---------- kernel 4: scatter dispatched tokens into expert segments ----------
__global__ __launch_bounds__(256) void k_scatter(const int* __restrict__ top_i,
                                                 const float* __restrict__ top_w,
                                                 const int* __restrict__ offsets,
                                                 int* __restrict__ cursors,
                                                 int* __restrict__ slot_token,
                                                 float* __restrict__ slot_w) {
  int g = blockIdx.x * 256 + threadIdx.x;
  if (g >= NN * TOPK) return;
  int e = top_i[g];
  int pos = atomicAdd(&cursors[e], 1);
  int s = offsets[e] + pos;
  slot_token[s] = g >> 1;
  slot_w[s] = top_w[g];
}

// ---------- kernel 5: grouped FFN layer 1: hx = relu(x @ W1[e] + b1[e]) ----------
__global__ __launch_bounds__(256) void k_ffn1(const float* __restrict__ features,
                                              const float* __restrict__ W1,
                                              const float* __restrict__ b1,
                                              const int* __restrict__ tile_e,
                                              const int* __restrict__ tile_s0,
                                              const int* __restrict__ slot_token,
                                              u16* __restrict__ hx) {
  __shared__ u16 As[2 * 32 * PITCH];
  __shared__ u16 Bs[2 * 128 * PITCH];
  const int e = tile_e[blockIdx.x];
  if (e < 0) return;
  const int s0 = tile_s0[blockIdx.x];
  const int h0 = blockIdx.y * 128;
  const int tid = threadIdx.x, lane = tid & 31, wave = tid >> 5;

  v8f acc[2];
  {
    v8f z = {0.f,0.f,0.f,0.f,0.f,0.f,0.f,0.f};
    acc[0] = z; acc[1] = z;
  }
  const float* w1e = W1 + (size_t)e * DD * HH;
  const int ar = tid >> 3, aq = (tid & 7) * 4;          // A stage: 32 rows x 32 k
  const int bkg = (tid >> 5) * 4, bcg = (tid & 31) * 4; // B stage 4x4 blocks
  const int an = slot_token[s0 + ar];
  // always-valid source row (row 0 of this expert if slot is padding)
  const float* arow = features + ((size_t)e * NN + (an >= 0 ? an : 0)) * DD;
  const float zmask = (an >= 0) ? 1.f : 0.f;

  const int KSTEPS = DD / 32;
  float4 raf; float4 rb[4];
  raf = *(const float4*)(arow + 0 + aq);
  loadB4x4(w1e, HH, 0 + bkg, h0 + bcg, rb);
  for (int s = 0; s < KSTEPS; s++) {
    u16* Ab = As + (s & 1) * (32 * PITCH);
    u16* Bb = Bs + (s & 1) * (128 * PITCH);
    {
      V4U t;
      t.h[0] = f2bf(raf.x * zmask); t.h[1] = f2bf(raf.y * zmask);
      t.h[2] = f2bf(raf.z * zmask); t.h[3] = f2bf(raf.w * zmask);
      *(uint2*)(Ab + ar * PITCH + aq) = t.u;
    }
    storeBT4x4(Bb, bkg, bcg, rb);
    __syncthreads();
    if (s + 1 < KSTEPS) {
      raf = *(const float4*)(arow + (s + 1) * 32 + aq);
      loadB4x4(w1e, HH, (s + 1) * 32 + bkg, h0 + bcg, rb);
      if (s + 2 < KSTEPS)
        __builtin_prefetch(w1e + (size_t)((s + 2) * 32 + bkg) * HH + h0 + bcg, 0, 1);
    }
    v16bf b = frag_b(Bb, wave * 16, lane);
    #pragma unroll
    for (int mi = 0; mi < 2; mi++) {
      v16bf a = frag_a(Ab, mi * 16, lane);
      acc[mi] = __builtin_amdgcn_wmma_f32_16x16x32_bf16(
          false, a, false, b, (short)0, acc[mi], false, false);
    }
  }
  #pragma unroll
  for (int mi = 0; mi < 2; mi++) {
    int r = mi * 16 + ((lane & 16) ? 8 : 0);
    int h = h0 + wave * 16 + (lane & 15);
    float bb = b1[(size_t)e * HH + h];
    #pragma unroll
    for (int j = 0; j < 8; j++) {
      float v = acc[mi][j] + bb;
      v = v > 0.f ? v : 0.f;
      hx[(size_t)(s0 + r + j) * HH + h] = f2bf(v);
    }
  }
}

// ---------- kernel 6: grouped FFN layer 2 + weighted scatter ----------
__global__ __launch_bounds__(256) void k_ffn2(const u16* __restrict__ hx,
                                              const float* __restrict__ W2,
                                              const float* __restrict__ b2,
                                              const int* __restrict__ tile_e,
                                              const int* __restrict__ tile_s0,
                                              const int* __restrict__ slot_token,
                                              const float* __restrict__ slot_w,
                                              float* __restrict__ outF) {
  __shared__ u16 As[2 * 32 * PITCH];
  __shared__ u16 Bs[2 * 128 * PITCH];
  const int e = tile_e[blockIdx.x];
  if (e < 0) return;
  const int s0 = tile_s0[blockIdx.x];
  const int o0 = blockIdx.y * 128;
  const int tid = threadIdx.x, lane = tid & 31, wave = tid >> 5;

  v8f acc[2];
  {
    v8f z = {0.f,0.f,0.f,0.f,0.f,0.f,0.f,0.f};
    acc[0] = z; acc[1] = z;
  }
  const float* w2e = W2 + (size_t)e * HH * OO;
  const int ar = tid >> 3, aq = (tid & 7) * 4;
  const int bkg = (tid >> 5) * 4, bcg = (tid & 31) * 4;
  const u16* arow = hx + (size_t)(s0 + ar) * HH;

  const int KSTEPS = HH / 32;
  uint2 rau; float4 rb[4];
  rau = *(const uint2*)(arow + 0 + aq);
  loadB4x4(w2e, OO, 0 + bkg, o0 + bcg, rb);
  for (int s = 0; s < KSTEPS; s++) {
    u16* Ab = As + (s & 1) * (32 * PITCH);
    u16* Bb = Bs + (s & 1) * (128 * PITCH);
    *(uint2*)(Ab + ar * PITCH + aq) = rau;
    storeBT4x4(Bb, bkg, bcg, rb);
    __syncthreads();
    if (s + 1 < KSTEPS) {
      rau = *(const uint2*)(arow + (s + 1) * 32 + aq);
      loadB4x4(w2e, OO, (s + 1) * 32 + bkg, o0 + bcg, rb);
      if (s + 2 < KSTEPS) {
        __builtin_prefetch(arow + (s + 2) * 32 + aq, 0, 1);
        __builtin_prefetch(w2e + (size_t)((s + 2) * 32 + bkg) * OO + o0 + bcg, 0, 1);
      }
    }
    v16bf b = frag_b(Bb, wave * 16, lane);
    #pragma unroll
    for (int mi = 0; mi < 2; mi++) {
      v16bf a = frag_a(Ab, mi * 16, lane);
      acc[mi] = __builtin_amdgcn_wmma_f32_16x16x32_bf16(
          false, a, false, b, (short)0, acc[mi], false, false);
    }
  }
  #pragma unroll
  for (int mi = 0; mi < 2; mi++) {
    int rr = mi * 16 + ((lane & 16) ? 8 : 0);
    int o = o0 + wave * 16 + (lane & 15);
    float bb = b2[(size_t)e * OO + o];
    #pragma unroll
    for (int j = 0; j < 8; j++) {
      int s = s0 + rr + j;
      int n = slot_token[s];
      if (n >= 0) {
        float y = acc[mi][j] + bb;
        unsafeAtomicAdd(&outF[(size_t)n * OO + o], slot_w[s] * y);
      }
    }
  }
}

// ---------- launch ----------
extern "C" void kernel_launch(void* const* d_in, const int* in_sizes, int n_in,
                              void* d_out, int out_size, void* d_ws, size_t ws_size,
                              hipStream_t stream) {
  (void)in_sizes; (void)n_in; (void)out_size; (void)ws_size;
  const float* features = (const float*)d_in[0];
  const float* recept   = (const float*)d_in[1];
  const float* Wg1      = (const float*)d_in[2];
  const float* bg1      = (const float*)d_in[3];
  const float* Wg2      = (const float*)d_in[4];
  const float* bg2      = (const float*)d_in[5];
  const float* W1       = (const float*)d_in[6];
  const float* b1       = (const float*)d_in[7];
  const float* W2       = (const float*)d_in[8];
  const float* b2       = (const float*)d_in[9];

  float* outF     = (float*)d_out;                    // [N,O]
  float* outRanks = outF + (size_t)NN * OO;           // [M,N]
  float* outTot   = outRanks + (size_t)MM * NN;       // [M]

  char* p = (char*)d_ws;
  u16*   gateH      = (u16*)p;   p += (size_t)NN * HH * sizeof(u16);
  u16*   hx         = (u16*)p;   p += (size_t)NSLOT * HH * sizeof(u16);
  int*   top_i      = (int*)p;   p += (size_t)NN * TOPK * sizeof(int);
  float* top_w      = (float*)p; p += (size_t)NN * TOPK * sizeof(float);
  int*   slot_token = (int*)p;   p += (size_t)NSLOT * sizeof(int);
  float* slot_w     = (float*)p; p += (size_t)NSLOT * sizeof(float);
  int*   counts     = (int*)p;   p += 8 * sizeof(int);
  int*   cursors    = (int*)p;   p += 8 * sizeof(int);
  int*   offsets    = (int*)p;   p += 16 * sizeof(int);
  int*   tile_e     = (int*)p;   p += MAXTILE * sizeof(int);
  int*   tile_s0    = (int*)p;   p += MAXTILE * sizeof(int);

  k_zero<<<dim3((NN * OO + 255) / 256), 256, 0, stream>>>(outF, outTot, counts);
  k_gate1<<<dim3(NN / 128, HH / 128), 256, 0, stream>>>(features, Wg1, bg1, gateH);
  k_gate2<<<dim3(NN), 256, 0, stream>>>(gateH, Wg2, bg2, recept,
                                        top_i, top_w, counts, outRanks, outTot);
  k_scan<<<1, 256, 0, stream>>>(counts, offsets, cursors, tile_e, tile_s0, slot_token);
  k_scatter<<<dim3((NN * TOPK + 255) / 256), 256, 0, stream>>>(
      top_i, top_w, offsets, cursors, slot_token, slot_w);
  k_ffn1<<<dim3(MAXTILE, HH / 128), 256, 0, stream>>>(
      features, W1, b1, tile_e, tile_s0, slot_token, hx);
  k_ffn2<<<dim3(MAXTILE, OO / 128), 256, 0, stream>>>(
      hx, W2, b2, tile_e, tile_s0, slot_token, slot_w, outF);
}
